// NeuralCA_73083163509121
// MI455X (gfx1250) — matrix operand order repaired and checked by reading, not compiled, not run on hardware
//
#include <hip/hip_runtime.h>

// CDNA5 / gfx1250 fused Neural-CA step.
// One 256-thread block (8 wave32) per 16x16 spatial tile:
//   async-stage 20x20x16 fp32 halo tile (GLOBAL_LOAD_ASYNC_TO_LDS_B128) +
//   WMMA-swizzled f16 weights into LDS, each wave pushes groups of 16 cells
//   through conv3x3->20 / 20->96 / 96->16 entirely with
//   v_wmma_f32_16x16x32_f16, then life-mask + store.

typedef __attribute__((ext_vector_type(16))) _Float16 v16h;
typedef __attribute__((ext_vector_type(8)))  float    v8f;
typedef __attribute__((ext_vector_type(4)))  int      v4i;

#define SZ    256
#define CH    16
#define PERC  20
#define FF    96
#define TS    16          // output tile
#define IR    20          // staged input region (halo 2)
#define DR    18          // dx region (halo 1, feeds post-life maxpool)
#define NCELL (DR * DR)   // 324
#define NGRP  21          // ceil(324/16)
#define NFRAG 19          // 10 (L1) + 6 (L2) + 3 (L3) B-fragments
#define ALIVE 0.1f

#if __has_builtin(__builtin_amdgcn_global_load_async_to_lds_b128)
#define NCA_ASYNC_STAGE 1
typedef __attribute__((address_space(1))) v4i as1_v4i;   // global int4
typedef __attribute__((address_space(3))) v4i as3_v4i;   // LDS int4
#endif

__device__ __forceinline__ void nca_wait_async() {
#if __has_builtin(__builtin_amdgcn_s_wait_asynccnt)
    __builtin_amdgcn_s_wait_asynccnt(0);
#else
    asm volatile("s_wait_asynccnt 0x0" ::: "memory");
#endif
}

__device__ __forceinline__ unsigned hash_u32(unsigned v) {
    v = v * 747796405u + 2891336453u;
    unsigned w = ((v >> ((v >> 28u) + 4u)) ^ v) * 277803737u;
    return (w >> 22u) ^ w;
}

__global__ __launch_bounds__(256) void nca_fused_kernel(
    const float* __restrict__ x,  const float* __restrict__ upd,
    const float* __restrict__ w1, const float* __restrict__ b1,
    const float* __restrict__ w2, const float* __restrict__ b2,
    const float* __restrict__ w3, const float* __restrict__ b3,
    float* __restrict__ out)
{
    __shared__ float                        s_in[IR * IR][CH];     // 25.6 KB
    __shared__ __align__(32) _Float16       s_w[NFRAG][32][16];    // 19.0 KB, B-fragment layout
    __shared__ float                        s_bias[132];           // b1|b2|b3
    __shared__ __align__(32) _Float16       s_h1[8][16][32];       // per-wave layer1 acts
    __shared__ __align__(32) _Float16       s_h2[8][16][FF];       // per-wave layer2 acts
    __shared__ float                        s_e[NCELL];            // updated energy ch on 18x18
    __shared__ float                        s_out[TS * TS][CH];    // updated central cells

    const int tid = threadIdx.x;
    const int blk = blockIdx.x;
    const int b   = blk >> 8;
    const int ty0 = ((blk >> 4) & 15) * TS;
    const int tx0 = (blk & 15) * TS;

    // ---- stage input halo tile: async global->LDS DMA, zero pad, electrode ----
#ifdef NCA_ASYNC_STAGE
    for (int i = tid; i < IR * IR; i += 256) {
        int iy = i / IR, ix = i % IR;
        int gy = ty0 + iy - 2, gx = tx0 + ix - 2;
        if ((unsigned)gy < SZ && (unsigned)gx < SZ) {
            as1_v4i* gsrc = (as1_v4i*)&x[(((size_t)b * SZ + gy) * SZ + gx) * CH];
            as3_v4i* ldst = (as3_v4i*)&s_in[i][0];
            // one address pair, 4 x 16B beats via the shared immediate offset
            __builtin_amdgcn_global_load_async_to_lds_b128(gsrc, ldst, 0,  0);
            __builtin_amdgcn_global_load_async_to_lds_b128(gsrc, ldst, 16, 0);
            __builtin_amdgcn_global_load_async_to_lds_b128(gsrc, ldst, 32, 0);
            __builtin_amdgcn_global_load_async_to_lds_b128(gsrc, ldst, 48, 0);
        }
    }
    nca_wait_async();               // ASYNCcnt==0 before patching / barrier
    for (int i = tid; i < IR * IR; i += 256) {
        int iy = i / IR, ix = i % IR;
        int gy = ty0 + iy - 2, gx = tx0 + ix - 2;
        if (!((unsigned)gy < SZ && (unsigned)gx < SZ)) {
            float4 z = {0.f, 0.f, 0.f, 0.f};
            float4* dst = (float4*)&s_in[i][0];
            dst[0] = z; dst[1] = z; dst[2] = z; dst[3] = z;
        } else if (gy == 0 && gx == 0) {
            s_in[i][0] = upd[b];    // in-electrode drive
        }
    }
#else
    for (int i = tid; i < IR * IR; i += 256) {
        int iy = i / IR, ix = i % IR;
        int gy = ty0 + iy - 2, gx = tx0 + ix - 2;
        float4* dst = (float4*)&s_in[i][0];
        if ((unsigned)gy < SZ && (unsigned)gx < SZ) {
            const float4* src = (const float4*)&x[(((size_t)b * SZ + gy) * SZ + gx) * CH];
            dst[0] = src[0]; dst[1] = src[1]; dst[2] = src[2]; dst[3] = src[3];
            if (gy == 0 && gx == 0) s_in[i][0] = upd[b];
        } else {
            float4 z = {0.f, 0.f, 0.f, 0.f};
            dst[0] = z; dst[1] = z; dst[2] = z; dst[3] = z;
        }
    }
#endif

    // ---- stage weights as f16 WMMA B-fragments (lane-swizzled) ----
    // B layout (16-bit, 32K x 16N): lane l: N = l&15; elem j<8: K=8h+j, j>=8: K=16+8h+(j-8).
    for (int t = tid; t < NFRAG * 32; t += 256) {
        int f = t >> 5, l = t & 31;
        int khi = (l >= 16) ? 8 : 0;
        _Float16* dst = &s_w[f][l][0];
        if (f < 10) {                       // layer1 conv im2col: K=tap*16+ch
            int kc = f >> 1, nt = f & 1;
            int n  = nt * 16 + (l & 15);
            for (int j = 0; j < 16; ++j) {
                int K = kc * 32 + khi + (j & 7) + ((j >> 3) << 4);
                float v = 0.f;
                if (K < 144 && n < PERC) v = w1[(size_t)K * PERC + n];
                dst[j] = (_Float16)v;
            }
        } else if (f < 16) {                // layer2: 20(->32) x 96
            int n = (f - 10) * 16 + (l & 15);
            for (int j = 0; j < 16; ++j) {
                int K = khi + (j & 7) + ((j >> 3) << 4);
                dst[j] = (_Float16)((K < PERC) ? w2[(size_t)K * FF + n] : 0.f);
            }
        } else {                            // layer3: 96 x 16
            int n = l & 15;
            for (int j = 0; j < 16; ++j) {
                int K = (f - 16) * 32 + khi + (j & 7) + ((j >> 3) << 4);
                dst[j] = (_Float16)w3[(size_t)K * CH + n];
            }
        }
    }
    if (tid < 132) {
        float v;
        if (tid < 20)       v = b1[tid];
        else if (tid < 116) v = b2[tid - 20];
        else                v = b3[tid - 116];
        s_bias[tid] = v;
    }
    __syncthreads();

    const int wid  = tid >> 5;
    const int lane = tid & 31;
    const int lh   = (lane >> 4) & 1;   // A: lanes>=16 hold K+8; C: rows M+8
    const int ln   = lane & 15;

    // ---- per-wave: groups of 16 cells through the 3-layer MLP with WMMA ----
    for (int g = wid; g < NGRP; g += 8) {
        int c  = g * 16 + ln;
        int cc = c < NCELL ? c : NCELL - 1;       // clamp pads of last group
        int ry = cc / DR, rx = cc % DR;           // cell in 18x18 region
        int cb = lh * 8;                          // channel sub-block for A

        // layer 1: conv3x3 as GEMM, K=144(->160), N=20(->32)
        v8f acc0 = {0.f,0.f,0.f,0.f,0.f,0.f,0.f,0.f};
        v8f acc1 = {0.f,0.f,0.f,0.f,0.f,0.f,0.f,0.f};
#pragma unroll
        for (int kc = 0; kc < 5; ++kc) {
            int tap0 = 2 * kc, tap1 = 2 * kc + 1;
            v16h a;
            {
                int ky = tap0 / 3, kx = tap0 % 3;
                const float* lo = &s_in[(ry + ky) * IR + (rx + kx)][cb];
#pragma unroll
                for (int i = 0; i < 8; ++i) a[i] = (_Float16)lo[i];
            }
            if (tap1 < 9) {
                int ky = tap1 / 3, kx = tap1 % 3;
                const float* hi = &s_in[(ry + ky) * IR + (rx + kx)][cb];
#pragma unroll
                for (int i = 0; i < 8; ++i) a[8 + i] = (_Float16)hi[i];
            } else {
#pragma unroll
                for (int i = 0; i < 8; ++i) a[8 + i] = (_Float16)0.f;
            }
            v16h bf0 = *(const v16h*)&s_w[2 * kc + 0][lane][0];
            v16h bf1 = *(const v16h*)&s_w[2 * kc + 1][lane][0];
            acc0 = __builtin_amdgcn_wmma_f32_16x16x32_f16(false, a, false, bf0,
                                                          (short)0, acc0, false, false);
            acc1 = __builtin_amdgcn_wmma_f32_16x16x32_f16(false, a, false, bf1,
                                                          (short)0, acc1, false, false);
        }
        {   // bias + relu -> per-wave LDS (C frag: N=ln fixed, M=lh*8+i)
            float bia0 = s_bias[ln];
            float bia1 = (16 + ln) < PERC ? s_bias[16 + ln] : 0.f;
#pragma unroll
            for (int i = 0; i < 8; ++i) {
                int m = lh * 8 + i;
                float h0 = acc0[i] + bia0; h0 = h0 > 0.f ? h0 : 0.f;
                float h1 = acc1[i] + bia1; h1 = h1 > 0.f ? h1 : 0.f;
                s_h1[wid][m][ln]      = (_Float16)h0;
                s_h1[wid][m][16 + ln] = (_Float16)h1;
            }
        }

        // layer 2: 20(->32) -> 96
        v16h a2;
        {
            const _Float16* lo = &s_h1[wid][ln][lh * 8];
            const _Float16* hi = &s_h1[wid][ln][16 + lh * 8];
#pragma unroll
            for (int i = 0; i < 8; ++i) { a2[i] = lo[i]; a2[8 + i] = hi[i]; }
        }
        v8f d[6];
#pragma unroll
        for (int nt = 0; nt < 6; ++nt) {
            v8f z = {0.f,0.f,0.f,0.f,0.f,0.f,0.f,0.f};
            v16h bf = *(const v16h*)&s_w[10 + nt][lane][0];
            d[nt] = __builtin_amdgcn_wmma_f32_16x16x32_f16(false, a2, false, bf,
                                                           (short)0, z, false, false);
        }
#pragma unroll
        for (int nt = 0; nt < 6; ++nt) {
            float bia = s_bias[20 + nt * 16 + ln];
#pragma unroll
            for (int i = 0; i < 8; ++i) {
                int m = lh * 8 + i;
                float h = d[nt][i] + bia; h = h > 0.f ? h : 0.f;
                s_h2[wid][m][nt * 16 + ln] = (_Float16)h;
            }
        }

        // layer 3: 96 -> 16
        v8f acc3 = {0.f,0.f,0.f,0.f,0.f,0.f,0.f,0.f};
#pragma unroll
        for (int kc = 0; kc < 3; ++kc) {
            v16h a3;
            const _Float16* lo = &s_h2[wid][ln][kc * 32 + lh * 8];
            const _Float16* hi = &s_h2[wid][ln][kc * 32 + 16 + lh * 8];
#pragma unroll
            for (int i = 0; i < 8; ++i) { a3[i] = lo[i]; a3[8 + i] = hi[i]; }
            v16h bf = *(const v16h*)&s_w[16 + kc][lane][0];
            acc3 = __builtin_amdgcn_wmma_f32_16x16x32_f16(false, a3, false, bf,
                                                          (short)0, acc3, false, false);
        }

        // epilogue: x_new = x + dx*fire; stash energy ring + central outputs
        {
            float bia = s_bias[116 + ln];       // b3[ch], ch = ln
#pragma unroll
            for (int i = 0; i < 8; ++i) {
                int m    = lh * 8 + i;
                int cell = g * 16 + m;
                if (cell < NCELL) {
                    int ryy = cell / DR, rxx = cell % DR;
                    int gy = ty0 + ryy - 1, gx = tx0 + rxx - 1;
                    float dxv  = acc3[i] + bia;
                    unsigned id = (unsigned)((b * SZ + gy) * SZ + gx);
                    float fire = (hash_u32(id ^ 0x9E3779B9u) & 1u) ? 1.f : 0.f;
                    float xold = s_in[(ryy + 1) * IR + (rxx + 1)][ln];
                    float xnew = xold + dxv * fire;
                    bool inimg = ((unsigned)gy < SZ) && ((unsigned)gx < SZ);
                    if (ln == 1) s_e[cell] = inimg ? xnew : 0.f;   // energy channel
                    if (ryy >= 1 && ryy <= TS && rxx >= 1 && rxx <= TS)
                        s_out[(ryy - 1) * TS + (rxx - 1)][ln] = xnew;
                }
            }
        }
    }
    __syncthreads();

    // ---- life masks (pre & post 3x3 maxpool on energy) + final store ----
    {
        int ty = tid >> 4, tx = tid & 15;
        float pre = -1e30f, post = -1e30f;
#pragma unroll
        for (int dy = -1; dy <= 1; ++dy)
#pragma unroll
            for (int dk = -1; dk <= 1; ++dk) {
                pre  = fmaxf(pre,  s_in[(ty + 2 + dy) * IR + (tx + 2 + dk)][1]);
                post = fmaxf(post, s_e[(ty + 1 + dy) * DR + (tx + 1 + dk)]);
            }
        float life = (pre > ALIVE && post > ALIVE) ? 1.f : 0.f;
        int gy = ty0 + ty, gx = tx0 + tx;
        const float* sp = &s_out[ty * TS + tx][0];
        float4 r[4];
#pragma unroll
        for (int q = 0; q < 4; ++q) {
            float4 v;
            v.x = sp[q * 4 + 0] * life; v.y = sp[q * 4 + 1] * life;
            v.z = sp[q * 4 + 2] * life; v.w = sp[q * 4 + 3] * life;
            r[q] = v;
        }
        if (gy == 0 && gx == 0) r[0].x = upd[b];   // re-drive electrode
        float4* o4 = (float4*)&out[(((size_t)b * SZ + gy) * SZ + gx) * CH];
        o4[0] = r[0]; o4[1] = r[1]; o4[2] = r[2]; o4[3] = r[3];
    }
}

extern "C" void kernel_launch(void* const* d_in, const int* in_sizes, int n_in,
                              void* d_out, int out_size, void* d_ws, size_t ws_size,
                              hipStream_t stream) {
    const float* x   = (const float*)d_in[0];
    const float* upd = (const float*)d_in[1];
    const float* w1  = (const float*)d_in[2];
    const float* b1  = (const float*)d_in[3];
    const float* w2  = (const float*)d_in[4];
    const float* b2  = (const float*)d_in[5];
    const float* w3  = (const float*)d_in[6];
    const float* b3  = (const float*)d_in[7];
    float* out = (float*)d_out;
    (void)in_sizes; (void)n_in; (void)out_size; (void)d_ws; (void)ws_size;

    dim3 grid(32 * (SZ / TS) * (SZ / TS));   // 8192 tiles
    dim3 block(256);                          // 8 wave32
    nca_fused_kernel<<<grid, block, 0, stream>>>(x, upd, w1, b1, w2, b2, w3, b3, out);
}
